// SelfAttention_38568806318843
// MI455X (gfx1250) — compile-verified, matrix operand order checked by reading, not run on hardware
//
#include <hip/hip_runtime.h>
#include <hip/hip_bf16.h>

typedef __bf16 bf16;
typedef __attribute__((ext_vector_type(4)))  __bf16 v4bf;
typedef __attribute__((ext_vector_type(8)))  __bf16 v8bf;
typedef __attribute__((ext_vector_type(16))) __bf16 v16bf;
typedef __attribute__((ext_vector_type(8)))  float  v8f;
typedef __attribute__((ext_vector_type(4)))  unsigned int u32x4;
typedef __attribute__((ext_vector_type(4)))  int i32x4;
typedef __attribute__((ext_vector_type(8)))  int i32x8;

#define B_DIM 8
#define S_DIM 2048
#define D_DIM 512
#define U_DIM 512

__device__ __forceinline__ v8f wmma_bf16(v16bf a, v16bf b, v8f c) {
    return __builtin_amdgcn_wmma_f32_16x16x32_bf16(
        /*neg_a=*/false, a, /*neg_b=*/false, b,
        /*c_mod=*/(short)0, c, /*reuse_a=*/false, /*reuse_b=*/false);
}

// Load a 16x32 bf16 WMMA operand fragment (A row-per-lane / B col-per-lane).
// Lane L holds row (L&15); K elements: [half*8 .. half*8+7] and [16+half*8 ..].
__device__ __forceinline__ v16bf load_frag(const bf16* rowbase, int pitch) {
    int lane = threadIdx.x & 31;
    int m = lane & 15;
    int h = lane >> 4;
    const bf16* p = rowbase + m * pitch + h * 8;
    v8bf lo = *(const v8bf*)(p);
    v8bf hi = *(const v8bf*)(p + 16);
    return __builtin_shufflevector(lo, hi, 0,1,2,3,4,5,6,7,8,9,10,11,12,13,14,15);
}

// ---------------------------------------------------------------------------
// TDM: DMA a 32x512 bf16 tile (rows contiguous, tensor_dim0_stride=512) from
// global memory into LDS with 16B padding every 1024B row -> LDS pitch 520
// bf16 elems, bank-conflict-free 16-lane fragment reads.
// D# per CDNA5 ISA ch.8: group0 = {count, lds_addr, global_addr, type=2},
// group1 = {data_size=2B, pad cfg, tensor/tile dims}.
// ---------------------------------------------------------------------------
__device__ __forceinline__ void tdm_load_k_tile(const bf16* gptr, unsigned lds_byte_off) {
    unsigned long long ga = (unsigned long long)(uintptr_t)gptr;
    u32x4 g0;
    g0[0] = 1u;                                            // count=1, user mode
    g0[1] = lds_byte_off;                                  // lds_addr
    g0[2] = (unsigned)(ga & 0xFFFFFFFFu);                  // global_addr[31:0]
    g0[3] = (unsigned)((ga >> 32) & 0x1FFFFFFu) | (2u << 30); // addr[56:32] | type=2
    i32x8 g1;
    g1[0] = (1 << 16)      // data_size = 2 bytes
          | (1 << 20)      // pad_enable
          | (7 << 22)      // pad_interval: every 256 DWORDs (= one 512-elem row)
          | (3 << 25);     // pad_amount: 4 DWORDs (16B) -> pitch 520 elems
    g1[1] = (512 << 16);   // tensor_dim0 = 512  (bits 79:48, low 16 here)
    g1[2] = (32 << 16);    // tensor_dim1 = 32   (bits 111:80, low 16 here)
    g1[3] = (512 << 16);   // tile_dim0 = 512    (bits 127:112)
    g1[4] = 32;            // tile_dim1 = 32     (bits 143:128)
    g1[5] = 512;           // tensor_dim0_stride = 512 (bits 207:160, low 32)
    g1[6] = 0;
    g1[7] = 0;
    i32x4 z4 = {};
#if defined(__clang_major__) && __clang_major__ >= 23
    i32x8 z8 = {};
    __builtin_amdgcn_tensor_load_to_lds(g0, g1, z4, z4, z8, 0);
#else
    __builtin_amdgcn_tensor_load_to_lds(g0, g1, z4, z4, 0);
#endif
}

// ---------------------------------------------------------------------------
// Kernel 1: QKV projection GEMM.  C[M=16384, N=512] = X[M,K=512] * W[K,N]
// fp32 in -> bf16 WMMA (f32 accum) -> bf16 out (feeds attention kernel).
// Block: 128 threads (4 waves), 64x64 tile, K-step 32.  gridDim.z selects W.
// ---------------------------------------------------------------------------
__global__ __launch_bounds__(128) void qkv_gemm_kernel(
    const float* __restrict__ X,
    const float* __restrict__ Wq, const float* __restrict__ Wk, const float* __restrict__ Wv,
    bf16* __restrict__ Qb, bf16* __restrict__ Kb, bf16* __restrict__ Vb)
{
    const int N = U_DIM, K = D_DIM;
    const int APITCH = 40;   // 32 + 8 pad (bf16 elems), rows 16B aligned
    __shared__ bf16 Al[64 * 40];   // [m][k]
    __shared__ bf16 Bl[64 * 40];   // transposed: [n][k]

    const int z = blockIdx.z;
    const float* W = (z == 0) ? Wq : ((z == 1) ? Wk : Wv);
    bf16* Out      = (z == 0) ? Qb : ((z == 1) ? Kb : Vb);

    const int n0 = blockIdx.x * 64;
    const int m0 = blockIdx.y * 64;
    const int t = threadIdx.x;
    const int lane = t & 31;
    const int w = t >> 5;
    const int h = lane >> 4;
    const int nl = lane & 15;

    v8f acc[4];
#pragma unroll
    for (int i = 0; i < 4; ++i) acc[i] = (v8f){};

    for (int kb = 0; kb < K; kb += 32) {
        __syncthreads();
        // A tile: 64x32 fp32 -> bf16 LDS (row-major)
#pragma unroll
        for (int i = 0; i < 4; ++i) {
            int idx = t + i * 128;
            int r = idx >> 3, c = (idx & 7) * 4;
            float4 f = *reinterpret_cast<const float4*>(X + (size_t)(m0 + r) * K + kb + c);
            v4bf o; o[0] = (bf16)f.x; o[1] = (bf16)f.y; o[2] = (bf16)f.z; o[3] = (bf16)f.w;
            *(v4bf*)(&Al[r * APITCH + c]) = o;
        }
        // B tile: 32x64 fp32 -> bf16 LDS transposed to [n][k]
#pragma unroll
        for (int i = 0; i < 4; ++i) {
            int idx = t + i * 128;
            int r = idx >> 4, c = (idx & 15) * 4;
            float4 f = *reinterpret_cast<const float4*>(W + (size_t)(kb + r) * N + n0 + c);
            Bl[(c + 0) * APITCH + r] = (bf16)f.x;
            Bl[(c + 1) * APITCH + r] = (bf16)f.y;
            Bl[(c + 2) * APITCH + r] = (bf16)f.z;
            Bl[(c + 3) * APITCH + r] = (bf16)f.w;
        }
        __syncthreads();

        v16bf a = load_frag(&Al[(w * 16) * APITCH], APITCH);
#pragma unroll
        for (int nt = 0; nt < 4; ++nt) {
            v16bf bb = load_frag(&Bl[(nt * 16) * APITCH], APITCH);
            acc[nt] = wmma_bf16(a, bb, acc[nt]);
        }
    }

    // C fragment: row = v + 8*half, col = lane&15
#pragma unroll
    for (int nt = 0; nt < 4; ++nt)
#pragma unroll
        for (int v = 0; v < 8; ++v) {
            int m = m0 + w * 16 + v + 8 * h;
            int n = n0 + nt * 16 + nl;
            Out[(size_t)m * N + n] = (bf16)acc[nt][v];
        }
}

// ---------------------------------------------------------------------------
// Kernel 2: flash attention over bf16 Q/K/V, fp32 accumulate + online softmax.
// Block: 128 threads (4 waves) handles 16 queries of one batch.
// Per 32-key step: TDM DMAs the K tile into LDS (padded pitch), all threads
// stage V transposed, waves 0/1 compute score tiles, softmax in LDS, all
// waves do P*V.  Dynamic LDS layout (94,464 B):
//   Qs  [16][520] bf16   @ 0
//   Ks  [32][520] bf16   @ 16640   (TDM destination, pad-generated pitch)
//   Vt  [512][40] bf16   @ 49920
//   Sts [16][33]  f32    @ 90880
//   Pt  [16][40]  bf16   @ 92992
//   mrow/lrow/arow [16]f @ 94272/94336/94400
// ---------------------------------------------------------------------------
#define SM_QS   0
#define SM_KS   16640
#define SM_VT   49920
#define SM_STS  90880
#define SM_PT   92992
#define SM_MR   94272
#define SM_LR   94336
#define SM_AR   94400
#define SM_TOTAL 94464

__global__ __launch_bounds__(128) void flash_attn_kernel(
    const bf16* __restrict__ Qb, const bf16* __restrict__ Kb,
    const bf16* __restrict__ Vb, float* __restrict__ out)
{
    const int S = S_DIM, U = U_DIM;
    const int QPITCH = 520;
    const int KPITCH = 520;  // produced by TDM pad (16B per 1024B row)
    const int VPITCH = 40;
    const float scale = 0.04419417382f;  // 1/sqrt(512)

    extern __shared__ __align__(16) char smem[];
    bf16*  Qs   = (bf16*)(smem + SM_QS);
    bf16*  Ks   = (bf16*)(smem + SM_KS);
    bf16*  Vt   = (bf16*)(smem + SM_VT);
    float* Sts  = (float*)(smem + SM_STS);
    bf16*  Pt   = (bf16*)(smem + SM_PT);
    float* mrow = (float*)(smem + SM_MR);
    float* lrow = (float*)(smem + SM_LR);
    float* arow = (float*)(smem + SM_AR);
    // LDS byte offset of Ks for the TDM descriptor: generic LDS addresses are
    // {shared_aperture_hi, offset32}, so truncation yields the LDS offset.
    const unsigned ks_lds_off = (unsigned)(uintptr_t)(void*)Ks;

    const int b  = blockIdx.y;
    const int q0 = blockIdx.x * 16;
    const int t = threadIdx.x;
    const int lane = t & 31;
    const int w = t >> 5;
    const int h = lane >> 4;
    const int nl = lane & 15;

    // Stage Q tile into LDS (bf16)
#pragma unroll
    for (int i = 0; i < 8; ++i) {
        int idx = t + i * 128;
        int r = idx >> 6, c = (idx & 63) * 8;
        *(v8bf*)(&Qs[r * QPITCH + c]) =
            *(const v8bf*)(Qb + ((size_t)(b * S + q0 + r)) * U + c);
    }
    if (t < 16) { mrow[t] = -1e30f; lrow[t] = 0.0f; }

    v8f acc[8];
#pragma unroll
    for (int i = 0; i < 8; ++i) acc[i] = (v8f){};
    __syncthreads();

    for (int kv0 = 0; kv0 < S; kv0 += 32) {
        // Async: TDM DMAs K tile [32 keys x 512 d] -> Ks (padded rows)
        if (w == 0)
            tdm_load_k_tile(Kb + ((size_t)(b * S + kv0)) * U, ks_lds_off);

        // Meanwhile all threads stage V tile transposed into LDS: Vt[u][key]
#pragma unroll
        for (int i = 0; i < 16; ++i) {
            int idx = t + i * 128;
            int key = idx >> 6, c = (idx & 63) * 8;
            v8bf vv = *(const v8bf*)(Vb + ((size_t)(b * S + kv0 + key)) * U + c);
#pragma unroll
            for (int j = 0; j < 8; ++j) Vt[(c + j) * VPITCH + key] = vv[j];
        }
        if (kv0 + 32 < S)
            __builtin_prefetch(Vb + ((size_t)(b * S + kv0 + 32 + (t & 31))) * U, 0, 0);

        if (w == 0) __builtin_amdgcn_s_wait_tensorcnt(0);
        __syncthreads();   // Ks (TDM) + Vt ready

        // Waves 0,1: score tiles (16q x 16k each) from LDS Q and LDS K
        if (w < 2) {
            v8f sacc = (v8f){};
            const bf16* kbase = &Ks[(w * 16) * KPITCH];
#pragma unroll
            for (int kb = 0; kb < U_DIM; kb += 32) {
                v16bf a  = load_frag(&Qs[kb], QPITCH);
                v16bf bb = load_frag(kbase + kb, KPITCH);
                sacc = wmma_bf16(a, bb, sacc);
            }
#pragma unroll
            for (int v = 0; v < 8; ++v)
                Sts[(v + 8 * h) * 33 + w * 16 + nl] = sacc[v] * scale;
        }
        __syncthreads();

        // Online softmax update: thread r<16 owns query row r
        if (t < 16) {
            float mold = mrow[t], mx = mold;
#pragma unroll
            for (int j = 0; j < 32; ++j) mx = fmaxf(mx, Sts[t * 33 + j]);
            float alpha = __expf(mold - mx);
            float sum = 0.0f;
#pragma unroll
            for (int j = 0; j < 32; ++j) {
                float p = __expf(Sts[t * 33 + j] - mx);
                sum += p;
                Pt[t * 40 + j] = (bf16)p;
            }
            mrow[t] = mx;
            arow[t] = alpha;
            lrow[t] = lrow[t] * alpha + sum;
        }
        __syncthreads();

        // All waves: rescale accumulators, then P(16x32) * V(32x16) per U tile
        float av[8];
#pragma unroll
        for (int v = 0; v < 8; ++v) av[v] = arow[v + 8 * h];
        v16bf a = load_frag(&Pt[0], 40);
#pragma unroll
        for (int ut = 0; ut < 8; ++ut) {
#pragma unroll
            for (int v = 0; v < 8; ++v) acc[ut][v] *= av[v];
            v16bf bb = load_frag(&Vt[(w * 128 + ut * 16) * VPITCH], VPITCH);
            acc[ut] = wmma_bf16(a, bb, acc[ut]);
        }
        __syncthreads();
    }

    // Normalize and write fp32 output
    float lv[8];
#pragma unroll
    for (int v = 0; v < 8; ++v) lv[v] = 1.0f / lrow[v + 8 * h];
#pragma unroll
    for (int ut = 0; ut < 8; ++ut)
#pragma unroll
        for (int v = 0; v < 8; ++v) {
            int q = q0 + v + 8 * h;
            int u = w * 128 + ut * 16 + nl;
            out[((size_t)(b * S + q)) * U + u] = acc[ut][v] * lv[v];
        }
}

extern "C" void kernel_launch(void* const* d_in, const int* in_sizes, int n_in,
                              void* d_out, int out_size, void* d_ws, size_t ws_size,
                              hipStream_t stream) {
    (void)in_sizes; (void)n_in; (void)out_size; (void)ws_size;
    const float* x  = (const float*)d_in[0];
    const float* Wq = (const float*)d_in[1];
    const float* Wk = (const float*)d_in[2];
    const float* Wv = (const float*)d_in[3];

    const size_t elems = (size_t)B_DIM * S_DIM * U_DIM;  // 8,388,608
    bf16* Qb = (bf16*)d_ws;
    bf16* Kb = Qb + elems;
    bf16* Vb = Kb + elems;

    // Kernel 1: Q,K,V projections (z = 0,1,2)
    dim3 g1(U_DIM / 64, (B_DIM * S_DIM) / 64, 3);
    qkv_gemm_kernel<<<g1, 128, 0, stream>>>(x, Wq, Wk, Wv, Qb, Kb, Vb);

    // Kernel 2: flash attention (dynamic LDS: Q + TDM K tile + V^T + softmax)
    dim3 g2(S_DIM / 16, B_DIM);
    flash_attn_kernel<<<g2, 128, SM_TOTAL, stream>>>(Qb, Kb, Vb, (float*)d_out);
}
